// StreamBranch1D_24318104830188
// MI455X (gfx1250) — compile-verified, matrix operand order checked by reading, not run on hardware
//
#include <hip/hip_runtime.h>
#include <math.h>

typedef __attribute__((ext_vector_type(16))) _Float16 v16h;
typedef __attribute__((ext_vector_type(8)))  float    v8f;

// Problem constants
#define B_   16
#define C_   22
#define L_   3000
#define D_   64
#define D2_  256
#define E_   8
#define K_   13
#define NPOS 7            // token positions 0..6 needed for n=0 output
#define KC   16384        // D2_*D_ : compress reduction length
#define NSPLIT 32         // split-K factor for the compress GEMM
#define KCH  (KC / NSPLIT)

__device__ __forceinline__ float gelu_f(float x) {
    return 0.5f * x * (1.0f + erff(x * 0.70710678118654752f));
}

// ---------------------------------------------------------------------------
// K1: tokenizer (conv1d stride2 pad6 -> GELU -> maxpool3 -> [cls; tokens] -> LN)
// Only conv positions 0..17 (pooled 0..5) feed token n=0's receptive field.
// One block per batch b. Writes H[b][n][d], n=0..6 (n=0 = LN'd cls token).
// ---------------------------------------------------------------------------
__global__ void k_tokenize(const float* __restrict__ x, const float* __restrict__ conv_w,
                           const float* __restrict__ conv_b, const float* __restrict__ cls,
                           const float* __restrict__ tok_g, const float* __restrict__ tok_b,
                           float* __restrict__ H) {
    __shared__ float convg[18 * 64];
    __shared__ float Hl[NPOS * 64];
    __shared__ float mn[NPOS], vr[NPOS];
    const int b = blockIdx.x;
    const int tid = threadIdx.x;

    for (int t = tid; t < 18 * 64; t += blockDim.x) {
        const int c = t >> 6, d = t & 63;
        float acc = conv_b[d];
        const int x0 = 2 * c - 6;
        for (int ch = 0; ch < C_; ++ch) {
            const float* xp = x + (size_t)b * C_ * L_ + (size_t)ch * L_;
            const float* wp = conv_w + d * (C_ * K_) + ch * K_;
            #pragma unroll
            for (int k = 0; k < K_; ++k) {
                const int xi = x0 + k;
                if (xi >= 0) acc = fmaf(wp[k], xp[xi], acc);   // xi < L_ always here
            }
        }
        convg[t] = gelu_f(acc);
    }
    __syncthreads();

    if (tid < 6 * 64) {                     // maxpool(3,3) -> positions 1..6
        const int p = tid >> 6, d = tid & 63;
        float m = fmaxf(convg[(3 * p) * 64 + d],
                  fmaxf(convg[(3 * p + 1) * 64 + d], convg[(3 * p + 2) * 64 + d]));
        Hl[(p + 1) * 64 + d] = m;
    } else if (tid < 7 * 64) {              // position 0 = cls token
        Hl[tid - 6 * 64] = cls[tid - 6 * 64];
    }
    __syncthreads();

    if (tid < NPOS) {                       // per-row LN stats over D=64
        float s = 0.f, s2 = 0.f;
        for (int d = 0; d < 64; ++d) { float v = Hl[tid * 64 + d]; s += v; s2 += v * v; }
        const float m = s * (1.f / 64.f);
        mn[tid] = m; vr[tid] = s2 * (1.f / 64.f) - m * m;
    }
    __syncthreads();

    if (tid < NPOS * 64) {
        const int n = tid >> 6, d = tid & 63;
        H[b * (NPOS * 64) + tid] =
            (Hl[tid] - mn[n]) * rsqrtf(vr[n] + 1e-5f) * tok_g[d] + tok_b[d];
    }
}

// ---------------------------------------------------------------------------
// K2: depthwise DS-conv at output position n=0 (only taps 6..12 in-bounds).
// Emits Z directly in f16 (WMMA A-operand format source).
// Zh[b][c2*64+d] = ds_conv_b[c2] + sum_{j=0..6} ds_conv_w[c2,6+j] * H[b][j][d]
// ---------------------------------------------------------------------------
__global__ void k_dsconv(const float* __restrict__ H, const float* __restrict__ ds_w,
                         const float* __restrict__ ds_b, _Float16* __restrict__ Zh) {
    const int idx = blockIdx.x * blockDim.x + threadIdx.x;   // < B_*KC
    if (idx >= B_ * KC) return;
    const int b = idx >> 14, k = idx & (KC - 1);
    const int c2 = k >> 6, d = k & 63;
    float acc = ds_b[c2];
    #pragma unroll
    for (int j = 0; j < 7; ++j)
        acc = fmaf(ds_w[c2 * K_ + 6 + j], H[b * (NPOS * 64) + j * 64 + d], acc);
    Zh[idx] = (_Float16)acc;
}

// ---------------------------------------------------------------------------
// K3: compress GEMM, split-K WMMA.  tokens_pre[16 x 256] = Z[16 x 16384] @ B,
// B[k][o] = ds_comp_w[o*16384 + k].  grid = (16 N-tiles, NSPLIT K-splits),
// one wave each -> 512 concurrent waves streaming the 16.8 MB weight once.
// Fragment layouts per ISA 7.12.2: lane<16 holds K in {0..7,16..23}, lanes
// 16..31 hold {8..15,24..31}; A lane->M, B lane->N, D elem r -> M=r+8*(lane>=16).
// ---------------------------------------------------------------------------
__global__ void k_compress_wmma(const _Float16* __restrict__ Zh, const float* __restrict__ comp,
                                float* __restrict__ partial) {
    const int lane  = threadIdx.x;           // 0..31, full wave, no divergence
    const int l15   = lane & 15;
    const int kbase = (lane >> 4) * 8;
    const int nGlob = blockIdx.x * 16 + l15;
    const int split = blockIdx.y;
    const _Float16* Arow = Zh   + (size_t)l15   * KC + (size_t)split * KCH;
    const float*    Brow = comp + (size_t)nGlob * KC + (size_t)split * KCH;
    v8f acc = {};
    for (int k0 = 0; k0 < KCH; k0 += 32) {
        if (k0 + 32 < KCH)
            __builtin_prefetch(Brow + k0 + 32 + kbase, 0, 1);   // global_prefetch_b8
        v16h a, bb;
        #pragma unroll
        for (int i = 0; i < 16; ++i) {
            const int koff = k0 + kbase + i + ((i >= 8) ? 8 : 0);
            a[i]  = Arow[koff];
            bb[i] = (_Float16)Brow[koff];
        }
        acc = __builtin_amdgcn_wmma_f32_16x16x32_f16(false, a, false, bb,
                                                     (short)0, acc, false, false);
    }
    #pragma unroll
    for (int r = 0; r < 8; ++r) {
        const int m = r + ((lane >> 4) * 8);            // batch row
        partial[((size_t)split * B_ + m) * D2_ + nGlob] = acc[r];
    }
}

// K3b: deterministic split-K reduction (+ compress bias)
__global__ void k_compress_reduce(const float* __restrict__ partial,
                                  const float* __restrict__ bias, float* __restrict__ tp) {
    const int b = blockIdx.x, f = threadIdx.x;
    float s = bias[f];
    for (int sp = 0; sp < NSPLIT; ++sp)
        s += partial[((size_t)sp * B_ + b) * D2_ + f];
    tp[b * D2_ + f] = s;
}

// ---------------------------------------------------------------------------
// K4: tokens = GELU(LN(tokens_pre; ds_g, ds_b)); writes f32 (for gating)
// and f16 (WMMA A-operand for the expert GEMMs).
// ---------------------------------------------------------------------------
__global__ void k_token_ln(const float* __restrict__ tp, const float* __restrict__ g,
                           const float* __restrict__ bta, float* __restrict__ tok,
                           _Float16* __restrict__ tokh) {
    __shared__ float red[256];
    const int b = blockIdx.x, f = threadIdx.x;
    const float x = tp[b * D2_ + f];
    red[f] = x; __syncthreads();
    for (int s = 128; s > 0; s >>= 1) { if (f < s) red[f] += red[f + s]; __syncthreads(); }
    const float m = red[0] * (1.f / 256.f); __syncthreads();
    red[f] = x * x; __syncthreads();
    for (int s = 128; s > 0; s >>= 1) { if (f < s) red[f] += red[f + s]; __syncthreads(); }
    const float v = red[0] * (1.f / 256.f) - m * m;
    const float t = gelu_f((x - m) * rsqrtf(v + 1e-5f) * g[f] + bta[f]);
    tok[b * D2_ + f]  = t;
    tokh[b * D2_ + f] = (_Float16)t;
}

// ---------------------------------------------------------------------------
// K5: expert + universal GEMMs with WMMA.
// grid = (16 N-tiles, 9): y<8 -> gelu(tok @ exp_w[y] + exp_b[y]),
//                         y==8 -> gelu(tok @ uni_w + uni_b)
// ---------------------------------------------------------------------------
__global__ void k_expert_wmma(const _Float16* __restrict__ tokh, const float* __restrict__ exp_w,
                              const float* __restrict__ exp_b, const float* __restrict__ uni_w,
                              const float* __restrict__ uni_b, float* __restrict__ expout,
                              float* __restrict__ uniout) {
    const int lane  = threadIdx.x;
    const int l15   = lane & 15;
    const int kbase = (lane >> 4) * 8;
    const int nGlob = blockIdx.x * 16 + l15;
    const int e     = blockIdx.y;
    const float* Bm   = (e < 8) ? (exp_w + (size_t)e * D2_ * D2_) : uni_w;   // [k][n], row stride 256
    const float* bias = (e < 8) ? (exp_b + e * D2_) : uni_b;
    const _Float16* Arow = tokh + (size_t)l15 * D2_;
    v8f acc = {};
    for (int k0 = 0; k0 < D2_; k0 += 32) {
        v16h a, bb;
        #pragma unroll
        for (int i = 0; i < 16; ++i) {
            const int koff = k0 + kbase + i + ((i >= 8) ? 8 : 0);
            a[i]  = Arow[koff];
            bb[i] = (_Float16)Bm[(size_t)koff * D2_ + nGlob];
        }
        acc = __builtin_amdgcn_wmma_f32_16x16x32_f16(false, a, false, bb,
                                                     (short)0, acc, false, false);
    }
    float* outp = (e < 8) ? (expout + (size_t)e * B_ * D2_) : uniout;
    #pragma unroll
    for (int r = 0; r < 8; ++r) {
        const int m = r + ((lane >> 4) * 8);
        outp[m * D2_ + nGlob] = gelu_f(acc[r] + bias[nGlob]);
    }
}

// ---------------------------------------------------------------------------
// K6: gating (top-2 softmax), combine with universal path, final LN -> d_out
// ---------------------------------------------------------------------------
__global__ void k_moe_final(const float* __restrict__ tok, const int* __restrict__ task_ids,
                            const float* __restrict__ task_embed, const float* __restrict__ gate_w,
                            const float* __restrict__ gate_b, const float* __restrict__ expout,
                            const float* __restrict__ uniout, const float* __restrict__ out_g,
                            const float* __restrict__ out_b, float* __restrict__ out) {
    __shared__ float lg[E_];
    __shared__ int   se1, se2;
    __shared__ float sg1, sg2, som;
    __shared__ float red[256];
    const int b = blockIdx.x, f = threadIdx.x;
    const int task = task_ids[b];

    if (f < E_) {
        float acc = gate_b[f];
        for (int i = 0; i < D2_; ++i) acc = fmaf(tok[b * D2_ + i], gate_w[i * E_ + f], acc);
        for (int i = 0; i < D2_; ++i)
            acc = fmaf(task_embed[task * D2_ + i], gate_w[(D2_ + i) * E_ + f], acc);
        lg[f] = acc;
    }
    __syncthreads();
    if (f == 0) {
        int e1 = 0; float v1 = lg[0];
        for (int e = 1; e < E_; ++e) if (lg[e] > v1) { v1 = lg[e]; e1 = e; }
        int e2 = (e1 == 0) ? 1 : 0; float v2 = lg[e2];
        for (int e = 0; e < E_; ++e)
            if (e != e1 && lg[e] > v2) { v2 = lg[e]; e2 = e; }
        const float g1 = 1.f / (1.f + expf(v2 - v1));   // softmax over top-2
        se1 = e1; se2 = e2; sg1 = g1; sg2 = 1.f - g1; som = 1.f - g1;
    }
    __syncthreads();

    const float y = sg1 * expout[(size_t)se1 * B_ * D2_ + b * D2_ + f]
                  + sg2 * expout[(size_t)se2 * B_ * D2_ + b * D2_ + f]
                  + som * uniout[b * D2_ + f];

    red[f] = y; __syncthreads();
    for (int s = 128; s > 0; s >>= 1) { if (f < s) red[f] += red[f + s]; __syncthreads(); }
    const float m = red[0] * (1.f / 256.f); __syncthreads();
    red[f] = y * y; __syncthreads();
    for (int s = 128; s > 0; s >>= 1) { if (f < s) red[f] += red[f + s]; __syncthreads(); }
    const float v = red[0] * (1.f / 256.f) - m * m;
    out[b * D2_ + f] = (y - m) * rsqrtf(v + 1e-5f) * out_g[f] + out_b[f];
}

// ---------------------------------------------------------------------------
extern "C" void kernel_launch(void* const* d_in, const int* in_sizes, int n_in,
                              void* d_out, int out_size, void* d_ws, size_t ws_size,
                              hipStream_t stream) {
    const float* x_stream   = (const float*)d_in[0];
    const int*   task_ids   = (const int*)  d_in[1];
    const float* conv_w     = (const float*)d_in[2];
    const float* conv_b     = (const float*)d_in[3];
    const float* cls_token  = (const float*)d_in[4];
    const float* tok_g      = (const float*)d_in[5];
    const float* tok_b      = (const float*)d_in[6];
    const float* ds_conv_w  = (const float*)d_in[7];
    const float* ds_conv_b  = (const float*)d_in[8];
    const float* ds_comp_w  = (const float*)d_in[9];
    const float* ds_comp_b  = (const float*)d_in[10];
    const float* ds_g       = (const float*)d_in[11];
    const float* ds_b       = (const float*)d_in[12];
    const float* task_embed = (const float*)d_in[13];
    const float* gate_w     = (const float*)d_in[14];
    const float* gate_b     = (const float*)d_in[15];
    const float* exp_w      = (const float*)d_in[16];
    const float* exp_b      = (const float*)d_in[17];
    const float* uni_w      = (const float*)d_in[18];
    const float* uni_b      = (const float*)d_in[19];
    const float* out_g      = (const float*)d_in[20];
    const float* out_b      = (const float*)d_in[21];
    float* out = (float*)d_out;

    // workspace layout (float-sized slots; halves carved out of float regions)
    float* ws    = (float*)d_ws;
    float*    H    = ws;                          // 16*448           =   7168 f
    _Float16* Zh   = (_Float16*)(H + B_ * NPOS * 64);   // 16*16384 h =  131072 f
    float*    part = (float*)(Zh + (size_t)B_ * KC);    // 32*16*256  =  131072 f
    float*    tp   = part + (size_t)NSPLIT * B_ * D2_;  // 16*256     =    4096 f
    float*    tok  = tp   + B_ * D2_;                   // 16*256     =    4096 f
    _Float16* tokh = (_Float16*)(tok + B_ * D2_);       // 16*256 h   =    2048 f
    float*    eo   = (float*)(tokh + (size_t)B_ * D2_); // 8*16*256   =   32768 f
    float*    uo   = eo + (size_t)E_ * B_ * D2_;        // 16*256     =    4096 f
    // total ~1.27 MB

    k_tokenize<<<B_, 512, 0, stream>>>(x_stream, conv_w, conv_b, cls_token, tok_g, tok_b, H);
    k_dsconv<<<(B_ * KC + 511) / 512, 512, 0, stream>>>(H, ds_conv_w, ds_conv_b, Zh);
    k_compress_wmma<<<dim3(D2_ / 16, NSPLIT), 32, 0, stream>>>(Zh, ds_comp_w, part);
    k_compress_reduce<<<B_, 256, 0, stream>>>(part, ds_comp_b, tp);
    k_token_ln<<<B_, 256, 0, stream>>>(tp, ds_g, ds_b, tok, tokh);
    k_expert_wmma<<<dim3(D2_ / 16, E_ + 1), 32, 0, stream>>>(tokh, exp_w, exp_b, uni_w, uni_b, eo, uo);
    k_moe_final<<<B_, 256, 0, stream>>>(tok, task_ids, task_embed, gate_w, gate_b,
                                        eo, uo, out_g, out_b, out);
    (void)in_sizes; (void)n_in; (void)out_size; (void)ws_size;
}
